// WaveNetVocoder_51556787421922
// MI455X (gfx1250) — matrix operand order repaired and checked by reading, not compiled, not run on hardware
//
#include <hip/hip_runtime.h>
#include <hip/hip_bf16.h>
#include <math.h>

// WaveNet vocoder: B=4, M=80, T=65536, R=D=S=64, L=20. kernel_size=1 =>
// whole net is a per-timestep MLP; all matmuls mapped to v_wmma_f32_16x16x32_bf16
// with fp32 accumulation held in registers across all 20 layers.

#define BB     4
#define MCH    80
#define TLEN   65536
#define NLAYER 20

// d_ws layout (bytes)
#define LAYERW_BYTES 24576                 // 3 matrices * 8 frags * 1KB (bf16, A-frag order)
#define OFF_LAYERW   0                     // 20 * 24576 = 491520
#define OFF_CAUSAL   491520                // 12 frags * 1KB (K padded 80->96)
#define OFF_POST1    503808                // 8 frags * 1KB
#define OFF_BSUM     512000                // 64 floats: sum_l b_skip[l]

typedef float        v8f   __attribute__((ext_vector_type(8)));
typedef __bf16       v16bf __attribute__((ext_vector_type(16)));
typedef unsigned int u32x4 __attribute__((ext_vector_type(4)));
typedef int          i32x8 __attribute__((ext_vector_type(8)));
typedef int          i32x4 __attribute__((ext_vector_type(4)));

__device__ __forceinline__ unsigned short f2bf(float f) {
  unsigned int u = __float_as_uint(f);
  u = u + 0x7FFFu + ((u >> 16) & 1u);
  return (unsigned short)(u >> 16);
}

// Pack two f32 -> bf16x2 in 3 VALU ops: round-half-up adds + v_perm_b32.
__device__ __forceinline__ unsigned int pkbf(float lo, float hi) {
  unsigned int ul = __float_as_uint(lo) + 0x8000u;
  unsigned int uh = __float_as_uint(hi) + 0x8000u;
  return __builtin_amdgcn_perm(uh, ul, 0x07060302u);  // {uh[31:16], ul[31:16]}
}

// Swap half-wave (lanes 0-15 <-> 16-31): ds_swizzle SWAPX16, one DS op.
__device__ __forceinline__ unsigned int swap16(unsigned int v) {
  return (unsigned int)__builtin_amdgcn_ds_swizzle((int)v, 0x401F);
}

__device__ __forceinline__ float ftanh(float x) {
#if __has_builtin(__builtin_amdgcn_tanhf)
  return __builtin_amdgcn_tanhf(x);          // v_tanh_f32 (TRANS, co-executes with WMMA)
#elif __has_builtin(__builtin_amdgcn_tanh_f32)
  return __builtin_amdgcn_tanh_f32(x);
#else
  float ax = fabsf(x);
  float e  = __expf(-2.f * ax);
  float r  = (1.f - e) * __builtin_amdgcn_rcpf(1.f + e);
  return copysignf(r, x);
#endif
}

// ---------------------------------------------------------------------------
// Prep kernel: pre-swizzle all weight matrices into exact WMMA A-fragment
// order (this code's contiguous K order, not the ISA table: lane l holds M=l&15; in-lane element p maps to
// K = (l>>4)*16 + p), so the main kernel's per-layer weight
// block is one contiguous 24KB blob streamed by the Tensor Data Mover.
// ---------------------------------------------------------------------------
__global__ void wavenet_prep(const float* __restrict__ w_causal,
                             const float* __restrict__ w_dil,
                             const float* __restrict__ w_res,
                             const float* __restrict__ w_skip,
                             const float* __restrict__ w_post1,
                             const float* __restrict__ b_skip,
                             unsigned char* __restrict__ ws) {
  const int gtid = blockIdx.x * blockDim.x + threadIdx.x;
  const int gsz  = gridDim.x * blockDim.x;

  unsigned short* lw = (unsigned short*)(ws + OFF_LAYERW);
  const int N1 = NLAYER * 3 * 8 * 512;
  for (int e = gtid; e < N1; e += gsz) {
    int q = e & 511, f = (e >> 9) & 7;
    int m3 = (e >> 12) % 3, l = e / (512 * 8 * 3);
    int kk = f >> 2, mt = f & 3;
    int lane = q >> 4, p = q & 15;
    int row = mt * 16 + (lane & 15);
    int col = kk * 32 + ((lane >> 4) << 4) + p;
    const float* src = (m3 == 0) ? w_dil : (m3 == 1) ? w_skip : w_res;
    lw[e] = f2bf(src[(l * 64 + row) * 64 + col]);
  }

  unsigned short* cw = (unsigned short*)(ws + OFF_CAUSAL);
  for (int e = gtid; e < 12 * 512; e += gsz) {
    int q = e & 511, f = e >> 9;
    int kk = f % 3, mt = f / 3;
    int lane = q >> 4, p = q & 15;
    int row = mt * 16 + (lane & 15);
    int col = kk * 32 + ((lane >> 4) << 4) + p;
    cw[e] = f2bf((col < MCH) ? w_causal[row * MCH + col] : 0.f);
  }

  unsigned short* pw = (unsigned short*)(ws + OFF_POST1);
  for (int e = gtid; e < 8 * 512; e += gsz) {
    int q = e & 511, f = e >> 9;
    int kk = f >> 2, mt = f & 3;
    int lane = q >> 4, p = q & 15;
    int row = mt * 16 + (lane & 15);
    int col = kk * 32 + ((lane >> 4) << 4) + p;
    pw[e] = f2bf(w_post1[row * 64 + col]);
  }

  float* bsum = (float*)(ws + OFF_BSUM);
  for (int i = gtid; i < 64; i += gsz) {
    float s = 0.f;
    for (int l = 0; l < NLAYER; ++l) s += b_skip[l * 64 + i];
    bsum[i] = s;
  }
}

// ---------------------------------------------------------------------------
// Main kernel helpers
// ---------------------------------------------------------------------------
__device__ __forceinline__ v8f wmma_bf16(v16bf a, v16bf b, v8f c) {
  return __builtin_amdgcn_wmma_f32_16x16x32_bf16(false, a, false, b, (short)0,
                                                 c, false, false);
}

union FragA { uint4 u[2]; v16bf v; };
union FragB { unsigned int w[8]; v16bf v; };

__device__ __forceinline__ v16bf load_afrag(const unsigned char* base, int lane) {
  const uint4* q = (const uint4*)(base + lane * 32);
  FragA r; r.u[0] = q[0]; r.u[1] = q[1];
  return r.v;
}

__device__ __forceinline__ v16bf as_bf(const unsigned int w[8]) {
  FragB r;
#pragma unroll
  for (int i = 0; i < 8; ++i) r.w[i] = w[i];
  return r.v;
}

// C/D fragment layout: lane l holds (M = mt*16 + j + (l>>4)*8, N = l&15).
// Bias depends on M only -> broadcast fragment straight from the [64] vector.
__device__ __forceinline__ v8f load_biasfrag(const float* base, int mt, int lane) {
  const float4* q = (const float4*)(base + mt * 16 + ((lane >> 4) << 3));
  float4 a = q[0], b = q[1];
  v8f r = {a.x, a.y, a.z, a.w, b.x, b.y, b.z, b.w};
  return r;
}

// Convert two fp32 C/D frags (row tiles 2kk, 2kk+1) into one bf16 B-operand
// frag (K-slice of 32). With this code's contiguous K order this is a pure half-wave
// permutation -> v_perm packs + ds_swizzle swaps, no LDS memory traffic.
__device__ __forceinline__ void cvt_pair(const v8f& clo, const v8f& chi,
                                         int lane, unsigned int out[8]) {
  const bool upper = (lane & 16) != 0;
#pragma unroll
  for (int v = 0; v < 4; ++v) {
    unsigned int lo  = pkbf(clo[2 * v], clo[2 * v + 1]);
    unsigned int hi  = pkbf(chi[2 * v], chi[2 * v + 1]);
    unsigned int xlo = swap16(lo);
    unsigned int xhi = swap16(hi);
    out[v]     = upper ? xhi : lo;
    out[v + 4] = upper ? hi  : xlo;
  }
}

// Tensor Data Mover: 1-D tile load of a contiguous bf16 blob into LDS.
// This toolchain's builtin is the 6-arg form:
//   (uint32x4 g0, int32x8 g1, int32x4 g2, int32x4 g3, int32x8 g4, i32 cpol)
__device__ __forceinline__ void tdm_load(void* dst_lds, const void* src, int nelem) {
  unsigned long long ga = (unsigned long long)(uintptr_t)src;
  u32x4 g0;
  g0[0] = 1u;                                              // count=1, user mode
  g0[1] = (unsigned int)(uintptr_t)dst_lds;                // lds_addr
  g0[2] = (unsigned int)ga;                                // global_addr[31:0]
  g0[3] = (unsigned int)((ga >> 32) & 0x1FFFFFFull) | (2u << 30);  // addr hi | type=2
  unsigned int n = (unsigned int)nelem;
  i32x8 g1;
  g1[0] = (int)(1u << 16);                      // wg_mask=0, data_size=1 (2B)
  g1[1] = (int)((n & 0xFFFFu) << 16);           // tensor_dim0[15:0]
  g1[2] = (int)((n >> 16) | (1u << 16));        // tensor_dim0[31:16] | tensor_dim1=1
  g1[3] = (int)((n & 0xFFFFu) << 16);           // tile_dim0
  g1[4] = 1;                                    // tile_dim1=1, tile_dim2=0
  g1[5] = (int)n;                               // tensor_dim0_stride[31:0]
  g1[6] = (int)((n & 0xFFFFu) << 16);           // dim0_stride hi=0 | dim1_stride[15:0]
  g1[7] = (int)(n >> 16);                       // dim1_stride[47:16]
  i32x4 z4 = {0, 0, 0, 0};
  i32x8 z8 = {0, 0, 0, 0, 0, 0, 0, 0};
  __builtin_amdgcn_tensor_load_to_lds(g0, g1, z4, z4, z8, 0);
}

// ---------------------------------------------------------------------------
// Main kernel: 256 threads = 8 waves; each wave owns 2 column tiles (32
// timesteps). h[64x32] and skip[64x32] live in fp32 accumulator frags for the
// whole network. Layer weights double-buffered in LDS via TDM.
// ---------------------------------------------------------------------------
__global__ __launch_bounds__(256, 1) void wavenet_main(
    const float* __restrict__ x, const unsigned char* __restrict__ ws,
    const float* __restrict__ b_causal, const float* __restrict__ b_dil,
    const float* __restrict__ b_res, const float* __restrict__ b_post1,
    const float* __restrict__ w_post2, const float* __restrict__ b_post2,
    float* __restrict__ out) {
  extern __shared__ unsigned char smem[];
  unsigned char* lds_w0 = smem;             // 24576
  unsigned char* lds_w1 = smem + 24576;     // 24576
  unsigned char* lds_x  = smem + 49152;     // 16 ctiles * 3 kk * 1KB = 49152

  const int tid  = threadIdx.x;
  const int lane = tid & 31;
  const int wave = tid >> 5;
  const int bidx = blockIdx.y;
  const int t0   = blockIdx.x * 256;

  const unsigned char* layerW  = ws + OFF_LAYERW;
  const unsigned char* causalW = ws + OFF_CAUSAL;
  const unsigned char* post1W  = ws + OFF_POST1;
  const float* bsum = (const float*)(ws + OFF_BSUM);

  if (wave == 0) tdm_load(lds_w0, layerW, LAYERW_BYTES / 2);  // layer 0 weights

  // Stage x tile (80 rows + 16 zero-pad) into bf16 B-fragment layout.
  {
    const int n = tid;                       // column 0..255 of this block
    const size_t xbase = (size_t)bidx * MCH * (size_t)TLEN + t0 + n;
    const int ct = n >> 4, nl = n & 15;
#pragma unroll 4
    for (int i = 0; i < 48; ++i) {
      int r = 2 * i;
      float v0 = (r < MCH) ? x[xbase + (size_t)r * TLEN] : 0.f;
      float v1 = (r + 1 < MCH) ? x[xbase + (size_t)(r + 1) * TLEN] : 0.f;
      int kk = r >> 5;
      int lp = (((r >> 4) & 1) << 4) | nl;
      int off = ((ct * 3 + kk) << 10) + lp * 32 + (r & 15) * 2;
      *(unsigned int*)(lds_x + off) = pkbf(v0, v1);
    }
  }
  __syncthreads();

  v8f h[4][2], skip[4][2];
  unsigned int hB[2][2][8];

  // Causal 1x1 conv: h = Wc @ x + bc  (K padded to 96 -> 3 K-steps)
#pragma unroll
  for (int mt = 0; mt < 4; ++mt) {
    v8f b = load_biasfrag(b_causal, mt, lane);
    h[mt][0] = b; h[mt][1] = b;
  }
#pragma unroll
  for (int kk = 0; kk < 3; ++kk) {
    v16bf a[4];
#pragma unroll
    for (int mt = 0; mt < 4; ++mt)
      a[mt] = load_afrag(causalW + (mt * 3 + kk) * 1024, lane);
#pragma unroll
    for (int c = 0; c < 2; ++c) {
      v16bf bb = load_afrag(lds_x + (((wave * 2 + c) * 3 + kk) << 10), lane);
#pragma unroll
      for (int mt = 0; mt < 4; ++mt) h[mt][c] = wmma_bf16(a[mt], bb, h[mt][c]);
    }
  }
#pragma unroll
  for (int c = 0; c < 2; ++c) {
    cvt_pair(h[0][c], h[1][c], lane, hB[c][0]);
    cvt_pair(h[2][c], h[3][c], lane, hB[c][1]);
  }
#pragma unroll
  for (int mt = 0; mt < 4; ++mt) {  // skip_sum starts at sum_l b_skip[l]
    v8f s0 = load_biasfrag(bsum, mt, lane);
    skip[mt][0] = s0; skip[mt][1] = s0;
  }

  for (int l = 0; l < NLAYER; ++l) {
    const unsigned char* wb = (l & 1) ? lds_w1 : lds_w0;
    unsigned char* wbn = (l & 1) ? lds_w0 : lds_w1;
    if (wave == 0) __builtin_amdgcn_s_wait_tensorcnt(0);
    __syncthreads();
    if (wave == 0 && l + 1 < NLAYER)
      tdm_load(wbn, layerW + (size_t)(l + 1) * LAYERW_BYTES, LAYERW_BYTES / 2);

    unsigned int fB[2][2][8];
#pragma unroll
    for (int c = 0; c < 2; ++c) {
      v8f f[4];
#pragma unroll
      for (int mt = 0; mt < 4; ++mt)
        f[mt] = load_biasfrag(b_dil + l * 64, mt, lane);   // C = bias
#pragma unroll
      for (int kk = 0; kk < 2; ++kk) {
        v16bf hb = as_bf(hB[c][kk]);
#pragma unroll
        for (int mt = 0; mt < 4; ++mt) {
          v16bf a = load_afrag(wb + ((kk * 4 + mt) << 10), lane);
          f[mt] = wmma_bf16(a, hb, f[mt]);
        }
      }
#pragma unroll
      for (int mt = 0; mt < 4; ++mt)
#pragma unroll
        for (int j = 0; j < 8; ++j) f[mt][j] = ftanh(f[mt][j]);
      cvt_pair(f[0], f[1], lane, fB[c][0]);
      cvt_pair(f[2], f[3], lane, fB[c][1]);
    }
    // skip += Ws @ f ; h += Wr @ f   (accumulate via WMMA C operand)
#pragma unroll
    for (int kk = 0; kk < 2; ++kk) {
      v16bf as[4], ar[4];
#pragma unroll
      for (int mt = 0; mt < 4; ++mt) {
        as[mt] = load_afrag(wb + 8192 + ((kk * 4 + mt) << 10), lane);
        ar[mt] = load_afrag(wb + 16384 + ((kk * 4 + mt) << 10), lane);
      }
#pragma unroll
      for (int c = 0; c < 2; ++c) {
        v16bf fb = as_bf(fB[c][kk]);
#pragma unroll
        for (int mt = 0; mt < 4; ++mt) {
          skip[mt][c] = wmma_bf16(as[mt], fb, skip[mt][c]);
          h[mt][c]   = wmma_bf16(ar[mt], fb, h[mt][c]);
        }
      }
    }
#pragma unroll
    for (int mt = 0; mt < 4; ++mt) {
      v8f br = load_biasfrag(b_res + l * 64, mt, lane);
      h[mt][0] += br; h[mt][1] += br;
    }
#pragma unroll
    for (int c = 0; c < 2; ++c) {
      cvt_pair(h[0][c], h[1][c], lane, hB[c][0]);
      cvt_pair(h[2][c], h[3][c], lane, hB[c][1]);
    }
  }

  // Post: s = tanh(skip); s2 = tanh(W1 @ s + b1); out = dot(w2, s2) + b2
  unsigned int sB[2][2][8];
#pragma unroll
  for (int c = 0; c < 2; ++c) {
#pragma unroll
    for (int mt = 0; mt < 4; ++mt)
#pragma unroll
      for (int j = 0; j < 8; ++j) skip[mt][c][j] = ftanh(skip[mt][c][j]);
    cvt_pair(skip[0][c], skip[1][c], lane, sB[c][0]);
    cvt_pair(skip[2][c], skip[3][c], lane, sB[c][1]);
  }
  v8f p1[4][2];
#pragma unroll
  for (int mt = 0; mt < 4; ++mt) {
    v8f b = load_biasfrag(b_post1, mt, lane);
    p1[mt][0] = b; p1[mt][1] = b;
  }
#pragma unroll
  for (int kk = 0; kk < 2; ++kk) {
    v16bf a[4];
#pragma unroll
    for (int mt = 0; mt < 4; ++mt)
      a[mt] = load_afrag(post1W + ((kk * 4 + mt) << 10), lane);
#pragma unroll
    for (int c = 0; c < 2; ++c) {
      v16bf sb = as_bf(sB[c][kk]);
#pragma unroll
      for (int mt = 0; mt < 4; ++mt) p1[mt][c] = wmma_bf16(a[mt], sb, p1[mt][c]);
    }
  }
  const float bp2 = b_post2[0];
#pragma unroll
  for (int c = 0; c < 2; ++c) {
    float acc = 0.f;
#pragma unroll
    for (int mt = 0; mt < 4; ++mt) {
      v8f w2 = load_biasfrag(w_post2, mt, lane);
#pragma unroll
      for (int j = 0; j < 8; ++j) acc += w2[j] * ftanh(p1[mt][c][j]);
    }
    // combine the two channel halves per column (lanes l and l^16)
    acc += __uint_as_float(swap16(__float_as_uint(acc)));
    if (lane < 16)
      out[(size_t)bidx * TLEN + t0 + (wave * 2 + c) * 16 + lane] = acc + bp2;
  }
}

extern "C" void kernel_launch(void* const* d_in, const int* in_sizes, int n_in,
                              void* d_out, int out_size, void* d_ws, size_t ws_size,
                              hipStream_t stream) {
  const float* x        = (const float*)d_in[0];
  const float* w_causal = (const float*)d_in[1];
  const float* b_causal = (const float*)d_in[2];
  const float* w_dil    = (const float*)d_in[3];
  const float* b_dil    = (const float*)d_in[4];
  const float* w_res    = (const float*)d_in[5];
  const float* b_res    = (const float*)d_in[6];
  const float* w_skip   = (const float*)d_in[7];
  const float* b_skip   = (const float*)d_in[8];
  const float* w_post1  = (const float*)d_in[9];
  const float* b_post1  = (const float*)d_in[10];
  const float* w_post2  = (const float*)d_in[11];
  const float* b_post2  = (const float*)d_in[12];
  unsigned char* ws = (unsigned char*)d_ws;
  float* outp = (float*)d_out;

  wavenet_prep<<<dim3(240), dim3(256), 0, stream>>>(
      w_causal, w_dil, w_res, w_skip, w_post1, b_skip, ws);
  wavenet_main<<<dim3(TLEN / 256, BB), dim3(256), 98304, stream>>>(
      x, ws, b_causal, b_dil, b_res, b_post1, w_post2, b_post2, outp);
  (void)in_sizes; (void)n_in; (void)out_size; (void)ws_size;
}
